// SparseMoELayer_30769145708829
// MI455X (gfx1250) — compile-verified, matrix operand order checked by reading, not run on hardware
//
#include <hip/hip_runtime.h>
#include <hip/hip_bf16.h>

// ---------------- problem constants (from reference) ----------------
#define NE      8        // num experts
#define DM      1024     // dim model
#define TOTAL   16384    // 8*2048 tokens
#define CAP     2560     // int(TOTAL/NE * 1.25)
#define NSLOT   (NE*CAP) // 20480 dispatched rows
#define NBLK    (TOTAL/256)

typedef __bf16 bf16;
typedef bf16  v8bf  __attribute__((ext_vector_type(8)));
typedef bf16  v16bf __attribute__((ext_vector_type(16)));
typedef float v8f   __attribute__((ext_vector_type(8)));

union Frag16 { v16bf v; v8bf h[2]; };

// gfx1250 Tensor Data Mover path (TENSORcnt-tracked). This toolchain exposes
// the 6-arg builtin form: (u32x4 g0, i32x8 g1, i32x4 g2, i32x4 g3, i32x8, cpol)
#if __has_builtin(__builtin_amdgcn_tensor_load_to_lds) && \
    __has_builtin(__builtin_amdgcn_s_wait_tensorcnt)
#define MOE_USE_TDM 1
typedef unsigned int u32x4 __attribute__((ext_vector_type(4)));
typedef int          i32x8 __attribute__((ext_vector_type(8)));
typedef int          i32x4 __attribute__((ext_vector_type(4)));
#else
#define MOE_USE_TDM 0
#endif

// ---------------- init: zero output, invalidate inverse map ----------------
__global__ void moe_init(float* out, int* s2t) {
    size_t idx = (size_t)blockIdx.x * 256 + threadIdx.x;
    if (idx < (size_t)TOTAL * DM + 1) out[idx] = 0.0f;
    if (idx < NSLOT) s2t[idx] = -1;
}

// ---------------- gating: logits -> softmax -> top1 ----------------
// one wave (32 lanes) per token; Wg staged in LDS (1024x8 f32 = 32KB)
__global__ void moe_gating(const float* __restrict__ x,
                           const float* __restrict__ Wg,
                           const float* __restrict__ bg,
                           int* __restrict__ eidx,
                           float* __restrict__ probs) {
    __shared__ float wgs[DM * NE];
    int tid = threadIdx.x;
    for (int i = tid; i < DM * NE; i += 256) wgs[i] = Wg[i];
    __syncthreads();

    int wave = tid >> 5, lane = tid & 31;
    int token = blockIdx.x * 8 + wave;
    const float* xr = x + (size_t)token * DM;

    float acc[NE];
#pragma unroll
    for (int e = 0; e < NE; ++e) acc[e] = 0.0f;
    for (int d = lane; d < DM; d += 32) {
        float xv = xr[d];
#pragma unroll
        for (int e = 0; e < NE; ++e) acc[e] += xv * wgs[d * NE + e];
    }
#pragma unroll
    for (int off = 16; off > 0; off >>= 1) {
#pragma unroll
        for (int e = 0; e < NE; ++e) acc[e] += __shfl_down(acc[e], off, 32);
    }
    if (lane == 0) {
        float logit[NE];
#pragma unroll
        for (int e = 0; e < NE; ++e) logit[e] = acc[e] + bg[e];
        float mx = logit[0]; int am = 0;
#pragma unroll
        for (int e = 1; e < NE; ++e)
            if (logit[e] > mx) { mx = logit[e]; am = e; }   // first-max tie break
        float s = 0.0f, p[NE];
#pragma unroll
        for (int e = 0; e < NE; ++e) { p[e] = __expf(logit[e] - mx); s += p[e]; }
        float inv = 1.0f / s;
        eidx[token] = am;
#pragma unroll
        for (int e = 0; e < NE; ++e) probs[(size_t)token * NE + e] = p[e] * inv;
    }
}

// ---------------- deterministic mean-prob reduction ----------------
__global__ void moe_prob_reduce(const float* __restrict__ probs, float* __restrict__ probSum) {
    __shared__ float part[256];
    int t = threadIdx.x, e = t & 7, g = t >> 3;     // 32 chunks x 8 experts
    float s = 0.0f;
    for (int i = 0; i < TOTAL / 32; ++i)            // 512 tokens per chunk, fixed order
        s += probs[(size_t)(g * (TOTAL / 32) + i) * NE + e];
    part[t] = s;
    __syncthreads();
    if (t < NE) {
        float a = 0.0f;
        for (int c = 0; c < 32; ++c) a += part[c * NE + t];
        probSum[t] = a;
    }
}

// ---------------- per-256-token-block histogram (integer atomics) ----------------
__global__ void moe_hist(const int* __restrict__ eidx, int* __restrict__ blockCnt) {
    __shared__ int cnt[NE];
    int t = threadIdx.x;
    if (t < NE) cnt[t] = 0;
    __syncthreads();
    atomicAdd(&cnt[eidx[blockIdx.x * 256 + t]], 1);
    __syncthreads();
    if (t < NE) blockCnt[blockIdx.x * NE + t] = cnt[t];
}

// ---------------- exclusive scan over blocks, per expert ----------------
__global__ void moe_scan(const int* __restrict__ blockCnt,
                         int* __restrict__ blockOff, int* __restrict__ totalCnt) {
    int e = threadIdx.x;
    if (e < NE) {
        int off = 0;
        for (int b = 0; b < NBLK; ++b) {
            blockOff[b * NE + e] = off;
            off += blockCnt[b * NE + e];
        }
        totalCnt[e] = off;
    }
}

// ---------------- in-block ordered rank -> slot + inverse map ----------------
__global__ void moe_rank(const int* __restrict__ eidx, const int* __restrict__ blockOff,
                         int* __restrict__ slotArr, int* __restrict__ s2t) {
    __shared__ int se[256];
    int t = threadIdx.x;
    int token = blockIdx.x * 256 + t;
    se[t] = eidx[token];
    __syncthreads();
    int e = se[t];
    int r = blockOff[blockIdx.x * NE + e];
    for (int j = 0; j < t; ++j) r += (se[j] == e) ? 1 : 0;   // flat-order rank
    int slot = (r < CAP) ? (e * CAP + r) : -1;
    slotArr[token] = slot;
    if (slot >= 0) s2t[slot] = token;
}

// ---------------- dispatch: gather token rows, convert to bf16 ----------------
__global__ void moe_dispatch(const float* __restrict__ x, const int* __restrict__ s2t,
                             bf16* __restrict__ Xbf) {
    int row = blockIdx.x;                       // [0, NSLOT)
    int t   = s2t[row];
    bf16* dst = Xbf + (size_t)row * DM;
    if (t >= 0) {
        const float* src = x + (size_t)t * DM;
        for (int d = threadIdx.x; d < DM; d += 256) dst[d] = (bf16)src[d];
    } else {
        for (int d = threadIdx.x; d < DM; d += 256) dst[d] = (bf16)0.0f;
    }
}

// ---------------- weight convert + transpose: W[e][k][n] -> Wt[e][n][k] (bf16) ----------------
__global__ void moe_wconv(const float* __restrict__ W, bf16* __restrict__ Wt) {
    __shared__ bf16 tile[32][33];
    int e = blockIdx.z;
    int k0 = blockIdx.x * 32, n0 = blockIdx.y * 32;
    const float* Wb = W + (size_t)e * DM * DM;
    for (int i = threadIdx.y; i < 32; i += 8)
        tile[i][threadIdx.x] = (bf16)Wb[(size_t)(k0 + i) * DM + n0 + threadIdx.x];
    __syncthreads();
    bf16* Wtb = Wt + (size_t)e * DM * DM;
    for (int i = threadIdx.y; i < 32; i += 8)
        Wtb[(size_t)(n0 + i) * DM + k0 + threadIdx.x] = tile[threadIdx.x][i];
}

#if MOE_USE_TDM
// Issue one 2D TDM tile load: rows x BK bf16 elements, row stride DM elements,
// compacted into LDS at ldsAddr. D# layout per CDNA5 ISA.
__device__ __forceinline__ void moe_tdm_tile(const bf16* gsrc, unsigned ldsAddr,
                                             int rows, int bk) {
    unsigned long long ga = (unsigned long long)(size_t)gsrc;
    u32x4 g0 = { 1u,                                   // count=1 (valid descriptor)
                 ldsAddr,                              // lds_addr [63:32]
                 (unsigned)ga,                         // global_addr lo
                 (unsigned)((ga >> 32) & 0x01FFFFFFu)  // global_addr hi (57-bit)
                   | 0x80000000u };                    // type=2 ("image") [127:126]
    i32x8 g1 = { 0x00010000,        // data_size=1 (2 bytes) [17:16]
                 bk   << 16,        // tensor_dim0 lo16  [63:48]
                 rows << 16,        // tensor_dim1 lo16  [95:80]
                 bk   << 16,        // tile_dim0         [127:112]
                 rows,              // tile_dim1         [143:128]
                 DM,                // tensor_dim0_stride lo32 [191:160]
                 0, 0 };
    i32x4 z4 = { 0, 0, 0, 0 };              // groups 2/3 unused for 2D tensors
    i32x8 z8 = { 0, 0, 0, 0, 0, 0, 0, 0 };
    __builtin_amdgcn_tensor_load_to_lds(g0, g1, z4, z4, z8, 0);
}
#endif

// ---------------- grouped GEMM via v_wmma_f32_16x16x32_bf16 ----------------
// C[e] = act(A[e][CAPxDM] @ Bt[e]^T + bias[e]);  Bt is [E][N][K] (pre-transposed)
// block: 256 thr = 8 waves (2 M x 4 N); block tile 128x128, wave tile 64x32,
// acc[4][2] = 64 VGPRs (spill-free geometry from round 1).
// TDM path: double-buffered LDS; wave 0 issues next tile's tensor_load_to_lds
// before the WMMAs and waits TENSORcnt after them -> DMA overlapped w/ math.
// MODE 0: ReLU -> bf16 H   |   MODE 1: scatter f32 rows to out via slot_to_token
template <int MODE>
__global__ void __launch_bounds__(256)
moe_gemm(const bf16* __restrict__ A, const bf16* __restrict__ Bt,
         const float* __restrict__ bias,
         bf16* __restrict__ Hout, float* __restrict__ Yout,
         const int* __restrict__ s2t) {
    constexpr int BM = 128, BN = 128, BK = 32;
#if MOE_USE_TDM
    __shared__ bf16 As[2][BM * BK];   // 2 x 8 KB
    __shared__ bf16 Bs[2][BN * BK];   // 2 x 8 KB
#else
    __shared__ bf16 As[1][BM * BK];
    __shared__ bf16 Bs[1][BN * BK];
#endif

    const int e  = blockIdx.z;
    const int m0 = blockIdx.x * BM;
    const int n0 = blockIdx.y * BN;
    const bf16* Ab = A  + ((size_t)e * CAP + m0) * DM;
    const bf16* Bb = Bt + (size_t)e * DM * DM + (size_t)n0 * DM;

    const int tid  = threadIdx.x;
    const int lane = tid & 31;
    const int wv   = tid >> 5;
    const int wm   = wv & 1;       // 2 wave rows  (64 M each)
    const int wn   = wv >> 1;      // 4 wave cols  (32 N each)
    const int lh   = lane >> 4;    // lane half: K-group select
    const int l16  = lane & 15;

    v8f acc[4][2] = {};
    int buf = 0;

#if MOE_USE_TDM
    // prologue: stage tile 0
    if (wv == 0) {
        moe_tdm_tile(Ab, (unsigned)(size_t)&As[0][0], BM, BK);
        moe_tdm_tile(Bb, (unsigned)(size_t)&Bs[0][0], BN, BK);
        __builtin_amdgcn_s_wait_tensorcnt(0);
    }
    __syncthreads();
#endif

    for (int kb = 0; kb < DM; kb += BK) {
        const int nxt = buf ^ 1;
#if MOE_USE_TDM
        if (wv == 0 && kb + BK < DM) {     // kick next tile's DMA (overlapped)
            moe_tdm_tile(Ab + kb + BK, (unsigned)(size_t)&As[nxt][0], BM, BK);
            moe_tdm_tile(Bb + kb + BK, (unsigned)(size_t)&Bs[nxt][0], BN, BK);
        }
#else
        // cooperative tile load: 128x32 bf16 each, 16B per slot, 512 slots
#pragma unroll
        for (int s = 0; s < 2; ++s) {
            int slot = tid + s * 256;
            int r = slot >> 2, c = (slot & 3) * 8;
            *(v8bf*)&As[0][r * BK + c] = *(const v8bf*)&Ab[(size_t)r * DM + kb + c];
            *(v8bf*)&Bs[0][r * BK + c] = *(const v8bf*)&Bb[(size_t)r * DM + kb + c];
        }
        if (kb + BK < DM) {
            __builtin_prefetch(&Ab[(size_t)(tid >> 2) * DM + kb + BK], 0, 3);
            __builtin_prefetch(&Bb[(size_t)(tid >> 2) * DM + kb + BK], 0, 3);
        }
        __syncthreads();
#endif

        // A fragments: lane l16 = M row; K per ISA 16-bit A layout:
        //   lanes 0-15: K {g*8..g*8+7, 16+g*8..}, g = lane>>4  (two ds_load_b128)
        Frag16 fa[4];
#pragma unroll
        for (int mf = 0; mf < 4; ++mf) {
            int row = wm * 64 + mf * 16 + l16;
            fa[mf].h[0] = *(const v8bf*)&As[buf][row * BK + lh * 8];
            fa[mf].h[1] = *(const v8bf*)&As[buf][row * BK + 16 + lh * 8];
        }
        // B fragments: lane l16 = N col (Bt row), same K striping as A
#pragma unroll
        for (int nf = 0; nf < 2; ++nf) {
            Frag16 fb;
            int n = wn * 32 + nf * 16 + l16;
            fb.h[0] = *(const v8bf*)&Bs[buf][n * BK + lh * 8];
            fb.h[1] = *(const v8bf*)&Bs[buf][n * BK + 16 + lh * 8];
#pragma unroll
            for (int mf = 0; mf < 4; ++mf)
                acc[mf][nf] = __builtin_amdgcn_wmma_f32_16x16x32_bf16(
                    false, fa[mf].v, false, fb.v,
                    (short)0, acc[mf][nf], false, false);
        }

#if MOE_USE_TDM
        if (wv == 0 && kb + BK < DM)
            __builtin_amdgcn_s_wait_tensorcnt(0);   // next tile landed
        __syncthreads();
        buf = nxt;
#else
        __syncthreads();
#endif
    }

    // epilogue; C/D layout: VGPR r -> M = r + (lane>>4)*8, N = lane&15
#pragma unroll
    for (int mf = 0; mf < 4; ++mf) {
#pragma unroll
        for (int nf = 0; nf < 2; ++nf) {
#pragma unroll
            for (int r = 0; r < 8; ++r) {
                int gm = m0 + wm * 64 + mf * 16 + lh * 8 + r;
                int gn = n0 + wn * 32 + nf * 16 + l16;
                float v = acc[mf][nf][r] + bias[e * DM + gn];
                if (MODE == 0) {
                    v = fmaxf(v, 0.0f);
                    Hout[((size_t)e * CAP + gm) * DM + gn] = (bf16)v;
                } else {
                    int t = s2t[e * CAP + gm];
                    if (t >= 0) Yout[(size_t)t * DM + gn] = v;
                }
            }
        }
    }
}

// ---------------- load-balancing loss ----------------
__global__ void moe_loss(const float* __restrict__ probSum, const int* __restrict__ totalCnt,
                         float* __restrict__ outLoss) {
    if (threadIdx.x == 0) {
        float acc = 0.0f;
        for (int e = 0; e < NE; ++e) {
            float pe = probSum[e] * (1.0f / TOTAL);
            float de = (float)totalCnt[e] * (1.0f / TOTAL);
            acc += pe * de * (float)(NE * NE);
        }
        outLoss[0] = acc * (1.0f / NE);
    }
}

// ---------------- launch ----------------
extern "C" void kernel_launch(void* const* d_in, const int* in_sizes, int n_in,
                              void* d_out, int out_size, void* d_ws, size_t ws_size,
                              hipStream_t stream) {
    const float* x  = (const float*)d_in[0];
    const float* Wg = (const float*)d_in[1];
    const float* bg = (const float*)d_in[2];
    const float* W1 = (const float*)d_in[3];
    const float* b1 = (const float*)d_in[4];
    const float* W2 = (const float*)d_in[5];
    const float* b2 = (const float*)d_in[6];
    float* out = (float*)d_out;

    char* ws = (char*)d_ws;
    auto carve = [&](size_t bytes) {
        char* p = ws;
        ws += (bytes + 255) & ~(size_t)255;
        return (void*)p;
    };
    int*   eidx     = (int*)  carve((size_t)TOTAL * 4);
    float* probs    = (float*)carve((size_t)TOTAL * NE * 4);
    float* probSum  = (float*)carve(NE * 4);
    int*   blockCnt = (int*)  carve((size_t)NBLK * NE * 4);
    int*   blockOff = (int*)  carve((size_t)NBLK * NE * 4);
    int*   totalCnt = (int*)  carve(NE * 4);
    int*   slotArr  = (int*)  carve((size_t)TOTAL * 4);
    int*   s2t      = (int*)  carve((size_t)NSLOT * 4);
    bf16*  Xbf      = (bf16*) carve((size_t)NSLOT * DM * 2);
    bf16*  Hbf      = (bf16*) carve((size_t)NSLOT * DM * 2);
    bf16*  W1t      = (bf16*) carve((size_t)NE * DM * DM * 2);
    bf16*  W2t      = (bf16*) carve((size_t)NE * DM * DM * 2);

    const size_t outN = (size_t)TOTAL * DM + 1;
    moe_init<<<(unsigned)((outN + 255) / 256), 256, 0, stream>>>(out, s2t);

    moe_gating<<<TOTAL / 8, 256, 0, stream>>>(x, Wg, bg, eidx, probs);
    moe_prob_reduce<<<1, 256, 0, stream>>>(probs, probSum);
    moe_hist<<<NBLK, 256, 0, stream>>>(eidx, blockCnt);
    moe_scan<<<1, 64, 0, stream>>>(blockCnt, blockOff, totalCnt);
    moe_rank<<<NBLK, 256, 0, stream>>>(eidx, blockOff, slotArr, s2t);
    moe_dispatch<<<NSLOT, 256, 0, stream>>>(x, s2t, Xbf);

    dim3 wb(32, 8);
    dim3 wg(DM / 32, DM / 32, NE);
    moe_wconv<<<wg, wb, 0, stream>>>(W1, W1t);
    moe_wconv<<<wg, wb, 0, stream>>>(W2, W2t);

    dim3 gg(CAP / 128, DM / 128, NE);
    moe_gemm<0><<<gg, 256, 0, stream>>>(Xbf, W1t, b1, Hbf, nullptr, s2t);
    moe_gemm<1><<<gg, 256, 0, stream>>>(Hbf, W2t, b2, nullptr, out, s2t);

    moe_loss<<<1, 32, 0, stream>>>(probSum, totalCnt, out + (size_t)TOTAL * DM);

    (void)in_sizes; (void)n_in; (void)out_size; (void)ws_size;
}
